// TinyAttentionMemory_67602785239363
// MI455X (gfx1250) — matrix operand order repaired and checked by reading, not compiled, not run on hardware
//
#include <hip/hip_runtime.h>
#include <hip/hip_bf16.h>

// MI455X / gfx1250, wave32. Conv2/Conv3 as implicit-GEMM via
// v_wmma_f32_16x16x32_f16 with LDS-staged, halo-padded channels-last f16
// activations (all inner-loop loads unconditional ds_load_b128).

typedef _Float16 half_t;
typedef __attribute__((ext_vector_type(16))) _Float16 v16h;
typedef __attribute__((ext_vector_type(8)))  _Float16 v8h;
typedef __attribute__((ext_vector_type(8)))  float    v8f;

#define B_   32
#define T_   16
#define HW0  128
#define HW1  64
#define HW2  32
#define C1   16
#define C2   32
#define C3   64
#define FEAT 128
#define DK   64
#define DV   128

#define SHUF16(lo, hi) __builtin_shufflevector(lo, hi, 0,1,2,3,4,5,6,7,8,9,10,11,12,13,14,15)

#define WMMA_F16(a, b, c) \
  __builtin_amdgcn_wmma_f32_16x16x32_f16(false, a, false, b, (short)0, c, false, false)

// ---------------------------------------------------------------------------
// Kernel 1: conv1(1->16,3x3,SAME) + ReLU + maxpool2.  Direct stencil (K=9 is
// too small for WMMA).  Output: act1[img][py][px][c16] f16 channels-last.
// ---------------------------------------------------------------------------
__global__ __launch_bounds__(256) void conv1_pool_kernel(
    const float* __restrict__ x, const float* __restrict__ w,
    const float* __restrict__ b, half_t* __restrict__ act1)
{
  __shared__ float sw[C1 * 9];
  __shared__ float sb[C1];
  int tid = threadIdx.x;
  if (tid < C1 * 9) sw[tid] = w[tid];
  if (tid < C1)     sb[tid] = b[tid];
  __syncthreads();

  int gid = blockIdx.x * 256 + tid;        // encodes ((img*64+py)*64+px)
  int px  = gid & 63;
  int py  = (gid >> 6) & 63;
  int img = gid >> 12;
  const float* xi = x + (size_t)img * HW0 * HW0;

  float p[4][4];
#pragma unroll
  for (int r = 0; r < 4; ++r) {
    int iy = 2 * py - 1 + r;
#pragma unroll
    for (int c = 0; c < 4; ++c) {
      int ix = 2 * px - 1 + c;
      p[r][c] = (iy >= 0 && iy < HW0 && ix >= 0 && ix < HW0) ? xi[iy * HW0 + ix] : 0.f;
    }
  }

  v16h out;
#pragma unroll
  for (int ch = 0; ch < C1; ++ch) {
    float a00 = 0.f, a01 = 0.f, a10 = 0.f, a11 = 0.f;
#pragma unroll
    for (int ky = 0; ky < 3; ++ky)
#pragma unroll
      for (int kx = 0; kx < 3; ++kx) {
        float wv = sw[ch * 9 + ky * 3 + kx];
        a00 += p[ky][kx] * wv;
        a01 += p[ky][kx + 1] * wv;
        a10 += p[ky + 1][kx] * wv;
        a11 += p[ky + 1][kx + 1] * wv;
      }
    float m = fmaxf(fmaxf(a00, a01), fmaxf(a10, a11)) + sb[ch];
    out[ch] = (half_t)fmaxf(m, 0.f);
  }
  *(v16h*)(act1 + (size_t)gid * 16) = out;   // 32B aligned store
}

// ---------------------------------------------------------------------------
// Kernel 2: conv2(16->32) + ReLU + maxpool2, implicit GEMM from LDS.
// K = (tap, ci) = 144 -> 5 chunks of 32 (last half zero at compile time).
// One block per image; 4 row-strips of 18 haloed act1 rows staged in LDS.
// 8 waves: wave = (ygroup<<1)|ntile; per strip each wave does 2 pooled rows.
// ---------------------------------------------------------------------------
__global__ __launch_bounds__(256) void conv2_pool_wmma(
    const half_t* __restrict__ act1, const float* __restrict__ w,
    const float* __restrict__ bias, half_t* __restrict__ act2)
{
  __shared__ __align__(32) half_t sa[18 * 66 * C1];  // haloed strip, 38,016 B
  __shared__ half_t sw[160 * C2];                    // [k][o] f16, 10,240 B
  __shared__ float  sb[C2];
  int tid = threadIdx.x;

  for (int i = tid; i < 160 * C2; i += 256) {
    int k = i >> 5, o = i & 31;
    half_t v = (half_t)0.f;
    if (k < 144) {
      int blk = k >> 4, ci = k & 15;
      int ky = blk / 3, kx = blk - ky * 3;
      v = (half_t)w[((o * C1 + ci) * 3 + ky) * 3 + kx];   // OIHW
    }
    sw[i] = v;
  }
  if (tid < C2) sb[tid] = bias[tid];
  // x-halo columns (persist across strips; rows are fully rewritten each strip)
  if (tid < 36) {
    int ry = tid >> 1, side = tid & 1;
    v16h z = {};
    *(v16h*)(sa + (ry * 66 + side * 65) * C1) = z;
  }

  int img = blockIdx.x;
  const half_t* g1 = act1 + (size_t)img * HW1 * HW1 * C1;
  half_t*       g2 = act2 + (size_t)img * HW2 * HW2 * C2;

  int lane  = tid & 31;
  int wv_   = tid >> 5;
  int nt    = wv_ & 1;                      // channel tile (0/1)
  int yg    = wv_ >> 1;                     // pooled-row subgroup (0..3)
  int laneL = lane & 15;
  int hi    = lane >> 4;                    // K-half selector
  int n     = nt * 16 + laneL;              // this lane's output channel
  int xm    = laneL;                        // lane's A-row x (plus xt*16)

  __syncthreads();

  // Preload B fragments: lane = N, element e <-> K = kc*32 + hi*8 + e + (e & 8)
  v16h bfrag[5];
#pragma unroll
  for (int kc = 0; kc < 5; ++kc)
#pragma unroll
    for (int e = 0; e < 16; ++e)
      bfrag[kc][e] = sw[(kc * 32 + hi * 8 + e + (e & 8)) * C2 + n];
  float bn = sb[n];

  for (int s = 0; s < 4; ++s) {
    // ---- stage strip: conv rows y = 16s-1 .. 16s+16 into sa rows 0..17 ----
    for (int c = tid; c < 18 * 128; c += 256) {
      int ry = c >> 7, cc = c & 127;
      int x = cc >> 1, hf = cc & 1;
      int y = 16 * s - 1 + ry;
      v8h r = {};
      if (y >= 0 && y < HW1)
        r = *(const v8h*)(g1 + ((y * HW1 + x) * C1 + hf * 8));
      *(v8h*)(sa + ((ry * 66 + x + 1) * C1 + hf * 8)) = r;
    }
    __syncthreads();

    // token prefetch of the next strip's leading rows (global_prefetch_b8)
    if (s < 3) {
      int ry = tid >> 7, cc = tid & 127;
      int x = cc >> 1, hf = cc & 1;
      int y = 16 * (s + 1) - 1 + ry;
      if (y >= 0 && y < HW1)
        __builtin_prefetch(g1 + ((y * HW1 + x) * C1 + hf * 8), 0, 1);
    }

    // ---- compute: pooled rows py = s*8 + (yg*2 .. yg*2+1) ----
#pragma unroll
    for (int pl = 0; pl < 2; ++pl) {
      int pl2 = yg * 2 + pl;                // local pooled row 0..7
      int py  = s * 8 + pl2;
#pragma unroll
      for (int xt = 0; xt < 4; ++xt) {
        int xb = xt * 16 + xm;              // haloed x base (ix = xb+kx-1 -> sa x = xb+kx)
        v8f c0 = {};
        v8f c1 = {};
#pragma unroll
        for (int kc = 0; kc < 5; ++kc) {
          v8h r00 = {}, r01 = {}, r10 = {}, r11 = {};
#pragma unroll
          for (int run = 0; run < 2; ++run) {
            const int blk = 2 * kc + run;   // compile-time after unroll
            if (blk < 9) {
              const int ky = blk / 3, kx = blk - 3 * (blk / 3);
              int base = ((2 * pl2 + ky) * 66 + xb + kx) * C1 + hi * 8;
              v8h ra = *(const v8h*)(sa + base);           // conv row y0
              v8h rb = *(const v8h*)(sa + base + 66 * C1); // conv row y0+1
              if (run == 0) { r00 = ra; r10 = rb; }
              else          { r01 = ra; r11 = rb; }
            }
          }
          v16h a0 = SHUF16(r00, r01);
          v16h a1 = SHUF16(r10, r11);
          c0 = WMMA_F16(a0, bfrag[kc], c0);
          c1 = WMMA_F16(a1, bfrag[kc], c1);
        }
        // bias + ReLU + 2x2 maxpool: C element e <-> M = e (+8 if hi)
#pragma unroll
        for (int j = 0; j < 4; ++j) {
          float m = fmaxf(fmaxf(c0[2 * j], c0[2 * j + 1]),
                          fmaxf(c1[2 * j], c1[2 * j + 1])) + bn;
          m = fmaxf(m, 0.f);
          int pxg = xt * 8 + hi * 4 + j;
          g2[(py * HW2 + pxg) * C2 + n] = (half_t)m;
        }
      }
    }
    __syncthreads();
  }
}

// ---------------------------------------------------------------------------
// Kernel 3: conv3(32->64) + ReLU + global mean, implicit GEMM from LDS.
// K = 288 = 9 chunks of 32 (exact).  One block per image; 4 row-strips of 10
// haloed act2 rows.  8 waves: wave = (yhalf<<2)|ntile.  Mean via ds_add_f32.
// ---------------------------------------------------------------------------
__global__ __launch_bounds__(256) void conv3_mean_wmma(
    const half_t* __restrict__ act2, const float* __restrict__ w,
    const float* __restrict__ bias, float* __restrict__ feat)
{
  __shared__ __align__(32) half_t sa[10 * 34 * C2];  // haloed strip, 21,760 B
  __shared__ half_t sw[288 * C3];                    // 36,864 B
  __shared__ float  sb[C3];
  __shared__ float  sfeat[C3];
  int tid = threadIdx.x;

  for (int i = tid; i < 288 * C3; i += 256) {
    int k = i >> 6, o = i & 63;
    int blk = k >> 5, ci = k & 31;
    int ky = blk / 3, kx = blk - ky * 3;
    sw[i] = (half_t)w[((o * C2 + ci) * 3 + ky) * 3 + kx];  // OIHW
  }
  if (tid < C3) { sb[tid] = bias[tid]; sfeat[tid] = 0.f; }
  if (tid < 20) {   // x-halo columns
    int ry = tid >> 1, side = tid & 1;
    v16h z = {};
    *(v16h*)(sa + (ry * 34 + side * 33) * C2) = z;
    *(v16h*)(sa + (ry * 34 + side * 33) * C2 + 16) = z;
  }

  int img = blockIdx.x;
  const half_t* g2 = act2 + (size_t)img * HW2 * HW2 * C2;

  int lane  = tid & 31;
  int wv_   = tid >> 5;
  int nt    = wv_ & 3;
  int yh    = wv_ >> 2;
  int laneL = lane & 15;
  int hi    = lane >> 4;
  int n     = nt * 16 + laneL;

  __syncthreads();

  v16h bfrag[9];
#pragma unroll
  for (int kc = 0; kc < 9; ++kc)
#pragma unroll
    for (int e = 0; e < 16; ++e)
      bfrag[kc][e] = sw[(kc * 32 + hi * 8 + e + (e & 8)) * C3 + n];
  float bn  = sb[n];
  float acc = 0.f;

  for (int s = 0; s < 4; ++s) {
    // ---- stage strip: conv rows y = 8s-1 .. 8s+8 into sa rows 0..9 ----
    for (int c = tid; c < 10 * 128; c += 256) {
      int ry = c >> 7, cc = c & 127;
      int x = cc >> 2, hf = cc & 3;
      int y = 8 * s - 1 + ry;
      v8h r = {};
      if (y >= 0 && y < HW2)
        r = *(const v8h*)(g2 + ((y * HW2 + x) * C2 + hf * 8));
      *(v8h*)(sa + ((ry * 34 + x + 1) * C2 + hf * 8)) = r;
    }
    __syncthreads();

    // ---- compute: conv rows y = 8s + (yh*4 .. yh*4+3) ----
#pragma unroll
    for (int lr4 = 0; lr4 < 4; ++lr4) {
      int lr = yh * 4 + lr4;                // local conv row 0..7 (sa row lr+1)
#pragma unroll
      for (int xt = 0; xt < 2; ++xt) {
        int xb = xt * 16 + laneL;
        v8f c = {};
#pragma unroll
        for (int kc = 0; kc < 9; ++kc) {
          const int ky = kc / 3, kx = kc - 3 * (kc / 3);
          int base = ((lr + ky) * 34 + xb + kx) * C2 + hi * 8;
          v8h r0 = *(const v8h*)(sa + base);        // K run 0 (ci = hi*8)
          v8h r1 = *(const v8h*)(sa + base + 16);   // K run 1 (ci = hi*8+16)
          v16h a = SHUF16(r0, r1);
          c = WMMA_F16(a, bfrag[kc], c);
        }
#pragma unroll
        for (int e = 0; e < 8; ++e) acc += fmaxf(c[e] + bn, 0.f);
      }
    }
    __syncthreads();
  }

  atomicAdd(&sfeat[n], acc);                // ds_add_f32
  __syncthreads();
  if (tid < C3) feat[(size_t)img * C3 + tid] = sfeat[tid] * (1.f / (HW2 * HW2));
}

// ---------------------------------------------------------------------------
// Head: fc, q/k/v projections, final-step attention over slots t=9..14.
// (lax.scan returns only the last carry; softmax is slot-order invariant.)
// ---------------------------------------------------------------------------
__global__ __launch_bounds__(256) void fc_kernel(
    const float* __restrict__ feat, const float* __restrict__ fcw,
    const float* __restrict__ fcb, float* __restrict__ f)
{
  int gid = blockIdx.x * 256 + threadIdx.x;   // 512*128
  int j = gid & 127, row = gid >> 7;
  const float* fr = feat + row * C3;
  const float* wr = fcw + j * C3;
  float s = fcb[j];
#pragma unroll
  for (int i = 0; i < C3; ++i) s += fr[i] * wr[i];
  f[gid] = s;
}

__global__ __launch_bounds__(256) void qkv_kernel(
    const float* __restrict__ f, const float* __restrict__ qw,
    const float* __restrict__ kw, const float* __restrict__ vw,
    float* __restrict__ q, float* __restrict__ k, float* __restrict__ v)
{
  int gid = blockIdx.x * 256 + threadIdx.x;   // 512*256
  int d = gid & 255, row = gid >> 8;
  const float* fr = f + row * FEAT;
  const float* wr;
  float* out;
  if (d < 64)       { wr = qw + d * FEAT;         out = q + row * DK + d; }
  else if (d < 128) { wr = kw + (d - 64) * FEAT;  out = k + row * DK + (d - 64); }
  else              { wr = vw + (d - 128) * FEAT; out = v + row * DV + (d - 128); }
  float s = 0.f;
#pragma unroll
  for (int i = 0; i < FEAT; ++i) s += fr[i] * wr[i];
  *out = s;
}

__global__ __launch_bounds__(128) void attn_kernel(
    const float* __restrict__ q, const float* __restrict__ k,
    const float* __restrict__ v, float* __restrict__ out)
{
  __shared__ float sc[6];
  int b = blockIdx.x, tid = threadIdx.x;
  const float* qt = q + ((size_t)b * T_ + (T_ - 1)) * DK;
  if (tid < 6) {
    int t = (T_ - 1) - 6 + tid;                 // frames 9..14
    const float* kt = k + ((size_t)b * T_ + t) * DK;
    float s = 0.f;
#pragma unroll
    for (int i = 0; i < DK; ++i) s += qt[i] * kt[i];
    sc[tid] = s * 0.125f;                       // 1/sqrt(64)
  }
  __syncthreads();
  float m = sc[0];
#pragma unroll
  for (int s = 1; s < 6; ++s) m = fmaxf(m, sc[s]);
  float e[6], sum = 0.f;
#pragma unroll
  for (int s = 0; s < 6; ++s) { e[s] = __expf(sc[s] - m); sum += e[s]; }
  float inv = 1.f / sum;
  float ctx = 0.f;
#pragma unroll
  for (int s = 0; s < 6; ++s)
    ctx += e[s] * inv * v[((size_t)b * T_ + (T_ - 7) + s) * DV + tid];
  out[(size_t)b * DV + tid] = ctx;
}

// ---------------------------------------------------------------------------
extern "C" void kernel_launch(void* const* d_in, const int* in_sizes, int n_in,
                              void* d_out, int out_size, void* d_ws, size_t ws_size,
                              hipStream_t stream)
{
  const float* bev = (const float*)d_in[0];
  const float* c1w = (const float*)d_in[1];
  const float* c1b = (const float*)d_in[2];
  const float* c2w = (const float*)d_in[3];
  const float* c2b = (const float*)d_in[4];
  const float* c3w = (const float*)d_in[5];
  const float* c3b = (const float*)d_in[6];
  const float* fcw = (const float*)d_in[7];
  const float* fcb = (const float*)d_in[8];
  const float* qw  = (const float*)d_in[9];
  const float* kw  = (const float*)d_in[10];
  const float* vw  = (const float*)d_in[11];
  float* outp = (float*)d_out;

  const int BT = B_ * T_;                         // 512 frames

  // Workspace carve-up (all offsets 256B aligned).
  char* ws = (char*)d_ws;
  size_t off = 0;
  half_t* act1 = (half_t*)(ws + off); off += (size_t)BT * HW1 * HW1 * C1 * sizeof(half_t); // 64 MB
  half_t* act2 = (half_t*)(ws + off); off += (size_t)BT * HW2 * HW2 * C2 * sizeof(half_t); // 32 MB
  float*  feat = (float*)(ws + off);  off += (size_t)BT * C3 * sizeof(float);
  float*  f    = (float*)(ws + off);  off += (size_t)BT * FEAT * sizeof(float);
  float*  qb   = (float*)(ws + off);  off += (size_t)BT * DK * sizeof(float);
  float*  kb   = (float*)(ws + off);  off += (size_t)BT * DK * sizeof(float);
  float*  vb   = (float*)(ws + off);  off += (size_t)BT * DV * sizeof(float);
  (void)ws_size; (void)in_sizes; (void)n_in; (void)out_size;

  conv1_pool_kernel<<<(BT * HW1 * HW1) / 256, 256, 0, stream>>>(bev, c1w, c1b, act1);
  conv2_pool_wmma  <<<BT, 256, 0, stream>>>(act1, c2w, c2b, act2);
  conv3_mean_wmma  <<<BT, 256, 0, stream>>>(act2, c3w, c3b, feat);
  fc_kernel        <<<(BT * FEAT) / 256, 256, 0, stream>>>(feat, fcw, fcb, f);
  qkv_kernel       <<<(BT * 256) / 256, 256, 0, stream>>>(f, qw, kw, vw, qb, kb, vb);
  attn_kernel      <<<B_, 128, 0, stream>>>(qb, kb, vb, outp);
}
